// SNN_64768106824233
// MI455X (gfx1250) — matrix operand order repaired and checked by reading, not compiled, not run on hardware
//
#include <hip/hip_runtime.h>

// ---------------------------------------------------------------------------
// SNN: fused (x @ W^T + b) GEMM + leaky-integrate-and-fire recurrence.
//   x: (B=512, T=12, D=1024) f32, W: (O=512, D=1024) f32, b: (O,) f32
//   out: (B, O) f32 = spike counts of last outer pass.
//
// Pipeline (fast path, needs ~27.3MB workspace):
//   1) cvt_split_kernel: one-time f32 -> bf16 hi/lo planes for x and W
//      (bf16x2 error split: val = hi + lo, |lo| ~ 2^-9 |val|).
//   2) snn_wmma_bf16_kernel: per-wave 16x16 (B x O) tile, 12 timestep
//      accumulators; hot loop is pure global_load_b128 + 3x
//      v_wmma_f32_16x16x32_bf16 per K-chunk (hi*hi + hi*lo + lo*hi),
//      then the 144-step LIF scan entirely in registers.
// Fallback (tiny ws): round-1 self-contained kernel with inline conversion.
// ---------------------------------------------------------------------------

typedef __attribute__((ext_vector_type(16))) __bf16 v16bf;
typedef __attribute__((ext_vector_type(8)))  __bf16 v8bf;
typedef __attribute__((ext_vector_type(8)))  float  v8f;

#define SNN_DECAY  0.2f
#define SNN_THRESH 0.5f

static __device__ __forceinline__ __bf16 f2bf(float f) {
    unsigned u = __builtin_bit_cast(unsigned, f);
    u += 0x7FFFu + ((u >> 16) & 1u);                 // round-to-nearest-even
    unsigned short h = (unsigned short)(u >> 16);
    return __builtin_bit_cast(__bf16, h);
}
static __device__ __forceinline__ float bf2f(__bf16 h) {
    unsigned u = ((unsigned)__builtin_bit_cast(unsigned short, h)) << 16;
    return __builtin_bit_cast(float, u);
}

// ---------------------------------------------------------------------------
// Pass 1: split f32 -> (bf16 hi, bf16 lo), 8 elements per thread.
// ---------------------------------------------------------------------------
__global__ __launch_bounds__(256) void cvt_split_kernel(
    const float* __restrict__ src, __bf16* __restrict__ hi,
    __bf16* __restrict__ lo, int n8)
{
    int i = blockIdx.x * 256 + threadIdx.x;
    if (i >= n8) return;
    const float4* p = (const float4*)src + (size_t)i * 2;
    float4 a = p[0], b = p[1];
    float f[8] = { a.x, a.y, a.z, a.w, b.x, b.y, b.z, b.w };
    v8bf h, l;
    #pragma unroll
    for (int j = 0; j < 8; ++j) {
        __bf16 hb = f2bf(f[j]);
        h[j] = hb;
        l[j] = f2bf(f[j] - bf2f(hb));
    }
    ((v8bf*)hi)[i] = h;
    ((v8bf*)lo)[i] = l;
}

// Fragment loaders (all 16B-aligned contiguous chunks -> global_load_b128).
static __device__ __forceinline__ v16bf load_afrag(const __bf16* p, int half) {
    // A 16x32, contiguous k order used here (not AMD's table order): lanes 0-15 K={0..15}, lanes 16-31 K={16..31}
    v8bf a = *(const v8bf*)(p + half * 16);
    v8bf b = *(const v8bf*)(p + half * 16 + 8);
    return __builtin_shufflevector(a, b, 0,1,2,3,4,5,6,7,8,9,10,11,12,13,14,15);
}
static __device__ __forceinline__ v16bf load_bfrag(const __bf16* p) {
    // B 32x16, same contiguous k order as A: per lane (N=lane&15) K=16*half..16*half+15
    v8bf a = *(const v8bf*)(p);
    v8bf b = *(const v8bf*)(p + 8);
    return __builtin_shufflevector(a, b, 0,1,2,3,4,5,6,7,8,9,10,11,12,13,14,15);
}

// ---------------------------------------------------------------------------
// Pass 2: fused GEMM (bf16x2 WMMA) + LIF recurrence.
// ---------------------------------------------------------------------------
__global__ __launch_bounds__(256) void snn_wmma_bf16_kernel(
    const __bf16* __restrict__ xh, const __bf16* __restrict__ xl,  // (B,T,D)
    const __bf16* __restrict__ wh, const __bf16* __restrict__ wl,  // (O,D)
    const float* __restrict__ bias, const int* __restrict__ twp,
    float* __restrict__ out)
{
    constexpr int T = 12, D = 1024, O = 512;

    const int lane = threadIdx.x & 31;
    const int wave = threadIdx.x >> 5;
    const int gw   = blockIdx.x * 8 + wave;   // 0..1023
    const int tM   = gw >> 5;                 // B-tile index, 0..31
    const int tN   = gw & 31;                 // O-tile index, 0..31
    const int n    = lane & 15;
    const int half = lane >> 4;

    const int rowB = tM * 16 + n;             // A row (M = lane&15)
    const int colO = tN * 16 + n;             // B col / output col (N = lane&15)

    const __bf16* xhB = xh + (size_t)rowB * (T * D);
    const __bf16* xlB = xl + (size_t)rowB * (T * D);
    const __bf16* whB = wh + (size_t)colO * D + half * 16;
    const __bf16* wlB = wl + (size_t)colO * D + half * 16;

    v8f acc[T] = {};

    #pragma unroll 1
    for (int k0 = 0; k0 < D; k0 += 32) {
        v16bf bh = load_bfrag(whB + k0);
        v16bf bl = load_bfrag(wlB + k0);

        if (k0 + 32 < D) {                    // prefetch next K-chunk
            __builtin_prefetch(whB + k0 + 32, 0, 3);
            __builtin_prefetch(xhB + k0 + 32, 0, 3);
        }

        #pragma unroll
        for (int t = 0; t < T; ++t) {
            v16bf ah = load_afrag(xhB + t * D + k0, half);
            v16bf al = load_afrag(xlB + t * D + k0, half);
            acc[t] = __builtin_amdgcn_wmma_f32_16x16x32_bf16(
                         false, ah, false, bh, (short)0, acc[t], false, false);
            acc[t] = __builtin_amdgcn_wmma_f32_16x16x32_bf16(
                         false, ah, false, bl, (short)0, acc[t], false, false);
            acc[t] = __builtin_amdgcn_wmma_f32_16x16x32_bf16(
                         false, al, false, bh, (short)0, acc[t], false, false);
        }
    }

    // ---- LIF recurrence in registers: element e -> (row tM*16+8*half+e, col colO)
    const float bia = bias[colO];
    const int   tw  = *twp;

    float v[8], s[8], sum[8];
    #pragma unroll
    for (int e = 0; e < 8; ++e) { v[e] = 0.0f; s[e] = 0.0f; sum[e] = 0.0f; }

    for (int it = 0; it < tw; ++it) {
        #pragma unroll
        for (int e = 0; e < 8; ++e) sum[e] = 0.0f;
        #pragma unroll
        for (int t = 0; t < T; ++t) {
            #pragma unroll
            for (int e = 0; e < 8; ++e) {
                float l = acc[t][e] + bia;
                v[e] = v[e] * SNN_DECAY * (1.0f - s[e]) + l;
                s[e] = (v[e] > SNN_THRESH) ? 1.0f : 0.0f;
                sum[e] += s[e];
            }
        }
    }

    #pragma unroll
    for (int e = 0; e < 8; ++e)
        out[(size_t)(tM * 16 + half * 8 + e) * O + colO] = sum[e];
}

// ---------------------------------------------------------------------------
// Fallback: self-contained (inline conversion) fused kernel, used only when
// workspace is too small for the bf16 planes.
// ---------------------------------------------------------------------------
__global__ __launch_bounds__(256) void snn_wmma_f32src_kernel(
    const float* __restrict__ x, const float* __restrict__ W,
    const float* __restrict__ bias, const int* __restrict__ twp,
    float* __restrict__ out)
{
    constexpr int T = 12, D = 1024, O = 512;
    const int lane = threadIdx.x & 31;
    const int wave = threadIdx.x >> 5;
    const int gw   = blockIdx.x * 8 + wave;
    const int tM   = gw >> 5;
    const int tN   = gw & 31;
    const int n    = lane & 15;
    const int half = lane >> 4;
    const int rowB = tM * 16 + n;
    const int colO = tN * 16 + n;

    v8f acc[T] = {};
    const float* wbase = W + (size_t)colO * D + half * 16;
    const float* xbase = x + (size_t)rowB * (T * D);

    #pragma unroll 1
    for (int k0 = 0; k0 < D; k0 += 32) {
        v16bf bh, bl;
        {
            const float4* p = (const float4*)(wbase + k0);
            float4 q0 = p[0], q1 = p[1], q2 = p[2], q3 = p[3];
            float fb[16] = { q0.x,q0.y,q0.z,q0.w, q1.x,q1.y,q1.z,q1.w,
                             q2.x,q2.y,q2.z,q2.w, q3.x,q3.y,q3.z,q3.w };
            #pragma unroll
            for (int i = 0; i < 16; ++i) {
                __bf16 h = f2bf(fb[i]);
                bh[i] = h;
                bl[i] = f2bf(fb[i] - bf2f(h));
            }
        }
        #pragma unroll
        for (int t = 0; t < T; ++t) {
            const float* xr = xbase + t * D + k0;
            const float4* pa = (const float4*)(xr + half * 16);
            const float4* pb = (const float4*)(xr + half * 16 + 8);
            float4 qa = pa[0], qb = pa[1], qc = pb[0], qd = pb[1];
            float fa[16] = { qa.x,qa.y,qa.z,qa.w, qb.x,qb.y,qb.z,qb.w,
                             qc.x,qc.y,qc.z,qc.w, qd.x,qd.y,qd.z,qd.w };
            v16bf ah, al;
            #pragma unroll
            for (int i = 0; i < 16; ++i) {
                __bf16 h = f2bf(fa[i]);
                ah[i] = h;
                al[i] = f2bf(fa[i] - bf2f(h));
            }
            acc[t] = __builtin_amdgcn_wmma_f32_16x16x32_bf16(
                         false, ah, false, bh, (short)0, acc[t], false, false);
            acc[t] = __builtin_amdgcn_wmma_f32_16x16x32_bf16(
                         false, ah, false, bl, (short)0, acc[t], false, false);
            acc[t] = __builtin_amdgcn_wmma_f32_16x16x32_bf16(
                         false, al, false, bh, (short)0, acc[t], false, false);
        }
    }

    const float bia = bias[colO];
    const int   tw  = *twp;
    float v[8], s[8], sum[8];
    #pragma unroll
    for (int e = 0; e < 8; ++e) { v[e] = 0.0f; s[e] = 0.0f; sum[e] = 0.0f; }
    for (int it = 0; it < tw; ++it) {
        #pragma unroll
        for (int e = 0; e < 8; ++e) sum[e] = 0.0f;
        #pragma unroll
        for (int t = 0; t < T; ++t) {
            #pragma unroll
            for (int e = 0; e < 8; ++e) {
                float l = acc[t][e] + bia;
                v[e] = v[e] * SNN_DECAY * (1.0f - s[e]) + l;
                s[e] = (v[e] > SNN_THRESH) ? 1.0f : 0.0f;
                sum[e] += s[e];
            }
        }
    }
    #pragma unroll
    for (int e = 0; e < 8; ++e)
        out[(size_t)(tM * 16 + half * 8 + e) * O + colO] = sum[e];
}

extern "C" void kernel_launch(void* const* d_in, const int* in_sizes, int n_in,
                              void* d_out, int out_size, void* d_ws, size_t ws_size,
                              hipStream_t stream) {
    (void)in_sizes; (void)n_in; (void)out_size;
    const float* x    = (const float*)d_in[0];
    const float* W    = (const float*)d_in[1];
    const float* bias = (const float*)d_in[2];
    const int*   tw   = (const int*)d_in[3];
    float*       out  = (float*)d_out;

    const size_t NX = (size_t)512 * 12 * 1024;   // x elements
    const size_t NW = (size_t)512 * 1024;        // W elements
    const size_t need = (NX * 2 + NW * 2) * 2;   // 4 bf16 planes, bytes

    if (ws_size >= need) {
        char* base = (char*)d_ws;
        __bf16* xh = (__bf16*)(base);
        __bf16* xl = (__bf16*)(base + NX * 2);
        __bf16* wh = (__bf16*)(base + NX * 4);
        __bf16* wl = (__bf16*)(base + NX * 4 + NW * 2);

        int nx8 = (int)(NX / 8), nw8 = (int)(NW / 8);
        cvt_split_kernel<<<dim3(nx8 / 256), dim3(256), 0, stream>>>(x, xh, xl, nx8);
        cvt_split_kernel<<<dim3(nw8 / 256), dim3(256), 0, stream>>>(W, wh, wl, nw8);
        snn_wmma_bf16_kernel<<<dim3(128), dim3(256), 0, stream>>>(
            xh, xl, wh, wl, bias, tw, out);
    } else {
        snn_wmma_f32src_kernel<<<dim3(128), dim3(256), 0, stream>>>(
            x, W, bias, tw, out);
    }
}
